// PaiNNLayer_63806034150131
// MI455X (gfx1250) — matrix operand order repaired and checked by reading, not compiled, not run on hardware
//
#include <hip/hip_runtime.h>
#include <hip/hip_bf16.h>

#define H 128
#define RBF 20

typedef __attribute__((ext_vector_type(16))) __bf16 v16bf;
typedef __attribute__((ext_vector_type(8)))  float  v8f;

// ---------- helpers ----------
static __device__ __forceinline__ unsigned short f2bf(float f) {
  unsigned int u = __builtin_bit_cast(unsigned int, f);
  u += 0x7FFFu + ((u >> 16) & 1u);          // round-to-nearest-even
  return (unsigned short)(u >> 16);
}
static __device__ __forceinline__ unsigned int pack2(float a, float b) {
  return (unsigned int)f2bf(a) | ((unsigned int)f2bf(b) << 16);
}
static __device__ __forceinline__ float silu(float x) { return x / (1.0f + __expf(-x)); }

static __device__ __forceinline__ v8f wmma_bf16(v16bf a, v16bf b, v8f c) {
  return __builtin_amdgcn_wmma_f32_16x16x32_bf16(false, a, false, b, (short)0, c, false, false);
}

// hardware fp32 atomic add (no-return, STOREcnt-tracked) -> global_atomic_add_f32
static __device__ __forceinline__ void gadd_f32(float* p, float x) {
  asm volatile("global_atomic_add_f32 %0, %1, off" :: "v"(p), "v"(x) : "memory");
}

// A-fragment (M=16 x K=32 bf16) per ISA layout from an LDS row-major tile.
// lane m0 = row, kh = lane>>4 selects {0..7,16..23} vs {8..15,24..31}.
static __device__ __forceinline__ v16bf load_a_frag(const unsigned short* xs, int m0, int kh,
                                                    int kbase, int stride) {
  union { unsigned int u[8]; v16bf h; } r;
  const unsigned int* p = (const unsigned int*)xs;
  int rowoff = m0 * stride;
#pragma unroll
  for (int i = 0; i < 8; ++i) {
    int k0 = kbase + ((i < 4) ? 0 : 16) + kh * 8 + (i & 3) * 2;
    r.u[i] = p[(rowoff + k0) >> 1];
  }
  return r.h;
}

// ---------- weight repack: row-major f32 [K x N] -> per-lane WMMA B fragments (bf16) ----------
// out element t: frag = t>>9, lane = (t>>4)&31, j = t&15 ; k = kt*32 + (j>>3)*16 + (lane>>4)*8 + (j&7)
__global__ void pack_frag(const float* __restrict__ W, unsigned short* __restrict__ out,
                          int Kreal, int Ncols, int NT, int total) {
  int t = blockIdx.x * blockDim.x + threadIdx.x;
  if (t >= total) return;
  int j = t & 15, lane = (t >> 4) & 31, frag = t >> 9;
  int kt = frag / NT, nt = frag % NT;
  int k = kt * 32 + (j >> 3) * 16 + (lane >> 4) * 8 + (j & 7);
  int n = nt * 16 + (lane & 15);
  float val = (k < Kreal) ? W[k * Ncols + n] : 0.0f;
  out[t] = f2bf(val);
}

// ---------- fused edge message kernel ----------
struct EdgeSmem {
  unsigned short x[16 * 160];   // [s[src] | rbf | pad] bf16, K padded 148->160
  unsigned short h[16 * 128];   // hidden after SiLU
  int src[16];
  int dst[16];
  float u[48];                  // edge_vec_unit, 16x3
};

__global__ void __launch_bounds__(128) painn_edge_kernel(
    const float* __restrict__ s, const float* __restrict__ v,
    const float* __restrict__ rbf, const float* __restrict__ evu,
    const float* __restrict__ mb1, const float* __restrict__ mb2,
    const int* __restrict__ ei,
    const unsigned short* __restrict__ wsW1, const unsigned short* __restrict__ wsW2,
    float* __restrict__ agg_s, float* __restrict__ agg_v, int E) {
  __shared__ EdgeSmem sm[4];
  int w = threadIdx.x >> 5, lane = threadIdx.x & 31;
  int base = (blockIdx.x * 4 + w) * 16;
  if (base >= E) return;
  int n0 = lane & 15, kh = lane >> 4;
  EdgeSmem& S = sm[w];

  if (lane < 16) S.src[lane] = ei[base + lane];
  else           S.dst[lane - 16] = ei[E + base + (lane - 16)];
  for (int t = lane; t < 48; t += 32) S.u[t] = evu[base * 3 + t];
  __builtin_amdgcn_wave_barrier();

  // stage X tile: gather s[src] (fp32->bf16) + rbf + zero pad
#pragma unroll 1
  for (int m = 0; m < 16; ++m) {
    int e = base + m;
    int sr = S.src[m];
    const float4 s4 = reinterpret_cast<const float4*>(s + (size_t)sr * H)[lane];
    unsigned int* xr = reinterpret_cast<unsigned int*>(&S.x[m * 160]);
    xr[lane * 2 + 0] = pack2(s4.x, s4.y);
    xr[lane * 2 + 1] = pack2(s4.z, s4.w);
    if (lane < 10) {
      float r0 = rbf[e * RBF + lane * 2 + 0];
      float r1 = rbf[e * RBF + lane * 2 + 1];
      xr[64 + lane] = pack2(r0, r1);
    } else if (lane < 16) {
      xr[64 + lane] = 0u;
    }
    __builtin_prefetch(v + (size_t)sr * (3 * H) + lane * 12, 0, 0);  // global_prefetch_b8
  }
  __builtin_amdgcn_wave_barrier();

  // GEMM1: [16x160] x [160x128] -> SiLU -> h (bf16 in LDS)
  v8f zero = {0.f, 0.f, 0.f, 0.f, 0.f, 0.f, 0.f, 0.f};
  v8f acc[8];
#pragma unroll
  for (int nt = 0; nt < 8; ++nt) acc[nt] = zero;
#pragma unroll
  for (int kt = 0; kt < 5; ++kt) {
    v16bf a = load_a_frag(S.x, n0, kh, kt * 32, 160);
#pragma unroll
    for (int nt = 0; nt < 8; ++nt) {
      v16bf b = *reinterpret_cast<const v16bf*>(wsW1 + ((kt * 8 + nt) * 512 + lane * 16));
      acc[nt] = wmma_bf16(a, b, acc[nt]);
    }
  }
#pragma unroll
  for (int nt = 0; nt < 8; ++nt) {
    int col = nt * 16 + n0;
    float bb = mb1[col];
#pragma unroll
    for (int r = 0; r < 8; ++r) {
      float z = acc[nt][r] + bb;
      S.h[(r + 8 * kh) * H + col] = f2bf(silu(z));
    }
  }
  __builtin_amdgcn_wave_barrier();

  // GEMM2 group a_ss: [16x128] x [128x128] + scatter-add to agg_s
  v8f accS[8];
#pragma unroll
  for (int nt = 0; nt < 8; ++nt) accS[nt] = zero;
#pragma unroll
  for (int kt = 0; kt < 4; ++kt) {
    v16bf a = load_a_frag(S.h, n0, kh, kt * 32, H);
#pragma unroll
    for (int nt = 0; nt < 8; ++nt) {
      v16bf b = *reinterpret_cast<const v16bf*>(wsW2 + ((kt * 24 + nt) * 512 + lane * 16));
      accS[nt] = wmma_bf16(a, b, accS[nt]);
    }
  }
#pragma unroll
  for (int nt = 0; nt < 8; ++nt) {
    int col = nt * 16 + n0;
    float bb = mb2[col];
#pragma unroll
    for (int r = 0; r < 8; ++r) {
      int m = r + 8 * kh;
      gadd_f32(agg_s + (size_t)S.dst[m] * H + col, accS[nt][r] + bb);
    }
  }

  // GEMM2 groups a_sv + a_vv together (one atomic per agg_v element)
  v8f accV[16];
#pragma unroll
  for (int i = 0; i < 16; ++i) accV[i] = zero;
#pragma unroll
  for (int kt = 0; kt < 4; ++kt) {
    v16bf a = load_a_frag(S.h, n0, kh, kt * 32, H);
#pragma unroll
    for (int nt = 0; nt < 8; ++nt) {
      v16bf b1 = *reinterpret_cast<const v16bf*>(wsW2 + ((kt * 24 + 8 + nt) * 512 + lane * 16));
      accV[nt] = wmma_bf16(a, b1, accV[nt]);
      v16bf b2 = *reinterpret_cast<const v16bf*>(wsW2 + ((kt * 24 + 16 + nt) * 512 + lane * 16));
      accV[8 + nt] = wmma_bf16(a, b2, accV[8 + nt]);
    }
  }
#pragma unroll
  for (int nt = 0; nt < 8; ++nt) {
    int col = nt * 16 + n0;
    float bs = mb2[H + col], bv = mb2[2 * H + col];
#pragma unroll
    for (int r = 0; r < 8; ++r) {
      int m = r + 8 * kh;
      float asv = accV[nt][r] + bs;
      float avv = accV[8 + nt][r] + bv;
      const float* vsrc = v + (size_t)S.src[m] * (3 * H) + col;
      float* ap = agg_v + (size_t)S.dst[m] * (3 * H) + col;
      float u0 = S.u[m * 3 + 0], u1 = S.u[m * 3 + 1], u2 = S.u[m * 3 + 2];
      gadd_f32(ap,         asv * u0 + avv * vsrc[0]);
      gadd_f32(ap + H,     asv * u1 + avv * vsrc[H]);
      gadd_f32(ap + 2 * H, asv * u2 + avv * vsrc[2 * H]);
    }
  }
}

// ---------- fused node update kernel (2 waves/block -> 41.7KB LDS) ----------
struct NodeSmem {
  unsigned short x[16 * 256];  // [s+agg_s | vnorm] bf16
  unsigned short h[16 * 128];
  float y[16 * 128];
  float red[64];
  float mu[16];
  float rs[16];
};

__global__ void __launch_bounds__(64) painn_node_kernel(
    const float* __restrict__ s, const float* __restrict__ v,
    const float* __restrict__ ub1, const float* __restrict__ ub2,
    const float* __restrict__ lng, const float* __restrict__ lnb,
    const unsigned short* __restrict__ wsU1, const unsigned short* __restrict__ wsU2,
    float* __restrict__ out_s /* aliases agg_s */, float* __restrict__ out_v /* aliases agg_v */,
    int N) {
  __shared__ NodeSmem sm[2];
  int w = threadIdx.x >> 5, lane = threadIdx.x & 31;
  int base = (blockIdx.x * 2 + w) * 16;
  if (base >= N) return;
  int n0 = lane & 15, kh = lane >> 4;
  NodeSmem& S = sm[w];

  // stage upd_in = [s+agg_s | ||v+agg_v||] as bf16
#pragma unroll 1
  for (int m = 0; m < 16; ++m) {
    int node = base + m;
    const float4 s4 = reinterpret_cast<const float4*>(s + (size_t)node * H)[lane];
    const float4 g4 = reinterpret_cast<const float4*>(out_s + (size_t)node * H)[lane];
    unsigned int* xr = reinterpret_cast<unsigned int*>(&S.x[m * 256]);
    xr[lane * 2 + 0] = pack2(s4.x + g4.x, s4.y + g4.y);
    xr[lane * 2 + 1] = pack2(s4.z + g4.z, s4.w + g4.w);
    float a0 = 0.f, a1 = 0.f, a2 = 0.f, a3 = 0.f;
#pragma unroll
    for (int d = 0; d < 3; ++d) {
      const float4 vv = reinterpret_cast<const float4*>(v + (size_t)node * (3 * H) + d * H)[lane];
      const float4 av = reinterpret_cast<const float4*>(out_v + (size_t)node * (3 * H) + d * H)[lane];
      float t0 = vv.x + av.x, t1 = vv.y + av.y, t2 = vv.z + av.z, t3 = vv.w + av.w;
      a0 += t0 * t0; a1 += t1 * t1; a2 += t2 * t2; a3 += t3 * t3;
    }
    xr[64 + lane * 2 + 0] = pack2(sqrtf(a0), sqrtf(a1));
    xr[64 + lane * 2 + 1] = pack2(sqrtf(a2), sqrtf(a3));
  }
  __builtin_amdgcn_wave_barrier();

  v8f zero = {0.f, 0.f, 0.f, 0.f, 0.f, 0.f, 0.f, 0.f};
  // GEMM U1: [16x256] x [256x128] -> SiLU
  v8f acc[8];
#pragma unroll
  for (int nt = 0; nt < 8; ++nt) acc[nt] = zero;
#pragma unroll
  for (int kt = 0; kt < 8; ++kt) {
    v16bf a = load_a_frag(S.x, n0, kh, kt * 32, 256);
#pragma unroll
    for (int nt = 0; nt < 8; ++nt) {
      v16bf b = *reinterpret_cast<const v16bf*>(wsU1 + ((kt * 8 + nt) * 512 + lane * 16));
      acc[nt] = wmma_bf16(a, b, acc[nt]);
    }
  }
#pragma unroll
  for (int nt = 0; nt < 8; ++nt) {
    int col = nt * 16 + n0;
    float bb = ub1[col];
#pragma unroll
    for (int r = 0; r < 8; ++r) {
      S.h[(r + 8 * kh) * H + col] = f2bf(silu(acc[nt][r] + bb));
    }
  }
  __builtin_amdgcn_wave_barrier();

  // GEMM U2: delta_s (nt 0..7) and gate_v (nt 8..15)
  v8f acc2[16];
#pragma unroll
  for (int i = 0; i < 16; ++i) acc2[i] = zero;
#pragma unroll
  for (int kt = 0; kt < 4; ++kt) {
    v16bf a = load_a_frag(S.h, n0, kh, kt * 32, H);
#pragma unroll
    for (int nt = 0; nt < 8; ++nt) {
      v16bf bd = *reinterpret_cast<const v16bf*>(wsU2 + ((kt * 16 + nt) * 512 + lane * 16));
      acc2[nt] = wmma_bf16(a, bd, acc2[nt]);
      v16bf bg = *reinterpret_cast<const v16bf*>(wsU2 + ((kt * 16 + 8 + nt) * 512 + lane * 16));
      acc2[8 + nt] = wmma_bf16(a, bg, acc2[8 + nt]);
    }
  }

  // y = s + delta_s -> LDS (fp32) for LayerNorm
#pragma unroll
  for (int nt = 0; nt < 8; ++nt) {
    int col = nt * 16 + n0;
    float bb = ub2[col];
#pragma unroll
    for (int r = 0; r < 8; ++r) {
      int m = r + 8 * kh;
      S.y[m * H + col] = s[(size_t)(base + m) * H + col] + acc2[nt][r] + bb;
    }
  }
  __builtin_amdgcn_wave_barrier();

  // LN stats: 2 lanes per row
  {
    int m = lane >> 1, h2 = lane & 1;
    const float4* yr = reinterpret_cast<const float4*>(&S.y[m * H + h2 * 64]);
    float sum = 0.f, sq = 0.f;
#pragma unroll
    for (int i = 0; i < 16; ++i) {
      float4 t = yr[i];
      sum += t.x + t.y + t.z + t.w;
      sq  += t.x * t.x + t.y * t.y + t.z * t.z + t.w * t.w;
    }
    S.red[lane * 2] = sum; S.red[lane * 2 + 1] = sq;
    __builtin_amdgcn_wave_barrier();
    sum += S.red[(lane ^ 1) * 2]; sq += S.red[(lane ^ 1) * 2 + 1];
    if (h2 == 0) {
      float mu = sum * (1.0f / H);
      float var = sq * (1.0f / H) - mu * mu;
      S.mu[m] = mu;
      S.rs[m] = rsqrtf(var + 1e-5f);
    }
  }
  __builtin_amdgcn_wave_barrier();

  // s_out (overwrites agg_s region — already consumed)
  const float4 g4 = reinterpret_cast<const float4*>(lng)[lane];
  const float4 b4 = reinterpret_cast<const float4*>(lnb)[lane];
#pragma unroll 1
  for (int m = 0; m < 16; ++m) {
    int node = base + m;
    float mu = S.mu[m], rs = S.rs[m];
    float4 yv = reinterpret_cast<const float4*>(&S.y[m * H])[lane];
    float4 o;
    o.x = (yv.x - mu) * rs * g4.x + b4.x;
    o.y = (yv.y - mu) * rs * g4.y + b4.y;
    o.z = (yv.z - mu) * rs * g4.z + b4.z;
    o.w = (yv.w - mu) * rs * g4.w + b4.w;
    reinterpret_cast<float4*>(out_s + (size_t)node * H)[lane] = o;
  }

  // v_out = gate_v * (v + agg_v), in-place over agg_v region (read-then-write per element)
#pragma unroll
  for (int nt = 0; nt < 8; ++nt) {
    int col = nt * 16 + n0;
    float gb = ub2[H + col];
#pragma unroll
    for (int r = 0; r < 8; ++r) {
      int m = r + 8 * kh, node = base + m;
      float gate = acc2[8 + nt][r] + gb;
      const float* vi = v + (size_t)node * (3 * H) + col;
      float* vo = out_v + (size_t)node * (3 * H) + col;
      vo[0]     = gate * (vi[0] + vo[0]);
      vo[H]     = gate * (vi[H] + vo[H]);
      vo[2 * H] = gate * (vi[2 * H] + vo[2 * H]);
    }
  }
}

// ---------- host launch ----------
extern "C" void kernel_launch(void* const* d_in, const int* in_sizes, int n_in,
                              void* d_out, int out_size, void* d_ws, size_t ws_size,
                              hipStream_t stream) {
  const float* s    = (const float*)d_in[0];
  const float* v    = (const float*)d_in[1];
  const float* rbf  = (const float*)d_in[2];
  const float* evu  = (const float*)d_in[3];
  const float* mW1  = (const float*)d_in[4];
  const float* mb1  = (const float*)d_in[5];
  const float* mW2  = (const float*)d_in[6];
  const float* mb2  = (const float*)d_in[7];
  const float* uW1  = (const float*)d_in[8];
  const float* ub1  = (const float*)d_in[9];
  const float* uW2  = (const float*)d_in[10];
  const float* ub2  = (const float*)d_in[11];
  const float* lng  = (const float*)d_in[12];
  const float* lnb  = (const float*)d_in[13];
  const int*   ei   = (const int*)d_in[14];

  const int N = in_sizes[0] / H;        // 50000
  const int E = in_sizes[14] / 2;       // 800000

  float* out   = (float*)d_out;
  float* agg_s = out;                   // [N,H]   aliases s_out
  float* agg_v = out + (size_t)N * H;   // [N,3,H] aliases v_out

  unsigned short* ws = (unsigned short*)d_ws;
  const int t1 = 5 * 8 * 512;    // W1 frags (K=160, NT=8)
  const int t2 = 4 * 24 * 512;   // W2 frags (K=128, NT=24)
  const int t3 = 8 * 8 * 512;    // U1 frags (K=256, NT=8)
  const int t4 = 4 * 16 * 512;   // U2 frags (K=128, NT=16)
  unsigned short* wsW1 = ws;
  unsigned short* wsW2 = ws + t1;
  unsigned short* wsU1 = ws + t1 + t2;
  unsigned short* wsU2 = ws + t1 + t2 + t3;

  // zero accumulators (also final-output buffer; fully overwritten by node kernel)
  hipMemsetAsync(d_out, 0, (size_t)out_size * sizeof(float), stream);

  pack_frag<<<t1 / 256, 256, 0, stream>>>(mW1, wsW1, H + RBF, H,     8,  t1);
  pack_frag<<<t2 / 256, 256, 0, stream>>>(mW2, wsW2, H,       3 * H, 24, t2);
  pack_frag<<<t3 / 256, 256, 0, stream>>>(uW1, wsU1, 2 * H,   H,     8,  t3);
  pack_frag<<<t4 / 256, 256, 0, stream>>>(uW2, wsU2, H,       2 * H, 16, t4);

  int edge_blocks = (E + 63) / 64;   // 4 waves x 16 edges per block
  painn_edge_kernel<<<edge_blocks, 128, 0, stream>>>(s, v, rbf, evu, mb1, mb2, ei,
                                                     wsW1, wsW2, agg_s, agg_v, E);

  int node_blocks = (N + 31) / 32;   // 2 waves x 16 nodes per block
  painn_node_kernel<<<node_blocks, 64, 0, stream>>>(s, v, ub1, ub2, lng, lnb,
                                                    wsU1, wsU2, agg_s, agg_v, N);
}
